// RBM_8701603742486
// MI455X (gfx1250) — matrix operand order repaired and checked by reading, not compiled, not run on hardware
//
#include <hip/hip_runtime.h>
#include <stdint.h>

// Problem constants (match reference)
#define K_      5
#define B_      1024
#define NH_     256
#define NV_     20000
#define SPLIT_  8            // split-K chunks for GEMM1
#define VSTEPS_ 625          // NV / 32
#define CHUNK_  79           // ceil(VSTEPS_/SPLIT_)

typedef __attribute__((ext_vector_type(16))) __bf16 v16bf;
typedef __attribute__((ext_vector_type(8)))  __bf16 v8bf;
typedef __attribute__((ext_vector_type(8)))  float  v8f;

// ---------------------------------------------------------------------------
// Deterministic stateless uniform(0,1) 32-bit hash (lowbias32-style) for the
// bernoulli samples. All flat indices fit in 32 bits (max ~102.4M < 2^31).
// Only cheap v_mul_lo_u32 / shifts -> ~3x less VALU than a 64-bit mix.
// ---------------------------------------------------------------------------
__device__ __forceinline__ float u01(unsigned idx, unsigned seed) {
    unsigned z = idx * 0x9E3779B9u + seed;
    z ^= z >> 16;  z *= 0x7FEB352Du;
    z ^= z >> 15;  z *= 0x846CA68Bu;
    z ^= z >> 16;
    return (float)(z >> 8) * (1.0f / 16777216.0f);
}

__device__ __forceinline__ v16bf cvt16(const float* f) {
    v16bf r;
#pragma unroll
    for (int j = 0; j < 16; ++j) r[j] = (__bf16)f[j];
    return r;
}

// ===========================================================================
// GEMM1: partial[s][b][h] = sum over this split-K chunk of
//        sum_k sum_v x[k,b,v] * W[k,h,v]     (NT gemm: both contiguous in v)
// Block: 256 thr = 8 waves (4Mx2N), block tile 128x128, wave tile 32x64.
// ===========================================================================
__global__ __launch_bounds__(256) void rbm_gemm1(
    const float* __restrict__ x, const float* __restrict__ W,
    float* __restrict__ partial) {
    const int bx = blockIdx.x;            // 0..7   (128 b-rows)
    const int by = blockIdx.y;            // 0..1   (128 h-cols)
    const int s  = blockIdx.z;            // 0..7   split-K chunk
    const int lane  = threadIdx.x & 31;
    const int wid   = threadIdx.x >> 5;
    const int waveM = wid >> 1;           // 0..3
    const int waveN = wid & 1;            // 0..1
    const int lo = lane & 15, hi = lane >> 4;

    v8f acc[2][4];
#pragma unroll
    for (int mi = 0; mi < 2; ++mi)
#pragma unroll
        for (int ni = 0; ni < 4; ++ni)
            acc[mi][ni] = (v8f){0.f,0.f,0.f,0.f,0.f,0.f,0.f,0.f};

    const int rBase = bx * 128 + waveM * 32;
    const int cBase = by * 128 + waveN * 64;
    const int st0 = s * CHUNK_;
    const int st1 = (st0 + CHUNK_ < VSTEPS_) ? (st0 + CHUNK_) : VSTEPS_;

    for (int k = 0; k < K_; ++k) {
        const float* xk = x + (size_t)k * B_  * NV_;
        const float* Wk = W + (size_t)k * NH_ * NV_;
        for (int st = st0; st < st1; ++st) {
            const int v0 = st * 32;
            // --- A fragments: x rows, 16x32 bf16 layout.
            // lane hi-half selects K = 16*hi + 0..15 contiguous
            v16bf afrag[2];
#pragma unroll
            for (int mi = 0; mi < 2; ++mi) {
                const float* src = xk + (size_t)(rBase + 16*mi + lo) * NV_ + v0 + 16*hi;
                float t[16];
                *(float4*)(t + 0)  = *(const float4*)(src + 0);
                *(float4*)(t + 4)  = *(const float4*)(src + 4);
                *(float4*)(t + 8)  = *(const float4*)(src + 8);
                *(float4*)(t + 12) = *(const float4*)(src + 12);
                __builtin_prefetch(src + 32, 0, 3);   // prefetch next v-step (near)
                afrag[mi] = cvt16(t);
            }
            // --- B fragments: W rows (col n = lo, K = 16*hi + 0..15 contiguous in v)
            v16bf bfrag[4];
#pragma unroll
            for (int ni = 0; ni < 4; ++ni) {
                const float* src = Wk + (size_t)(cBase + 16*ni + lo) * NV_ + v0 + 16*hi;
                float t[16];
                *(float4*)(t + 0)  = *(const float4*)(src + 0);
                *(float4*)(t + 4)  = *(const float4*)(src + 4);
                *(float4*)(t + 8)  = *(const float4*)(src + 8);
                *(float4*)(t + 12) = *(const float4*)(src + 12);
                __builtin_prefetch(src + 32, 0, 3);
                bfrag[ni] = cvt16(t);
            }
#pragma unroll
            for (int mi = 0; mi < 2; ++mi)
#pragma unroll
                for (int ni = 0; ni < 4; ++ni)
                    acc[mi][ni] = __builtin_amdgcn_wmma_f32_16x16x32_bf16(
                        false, afrag[mi], false, bfrag[ni],
                        (short)0, acc[mi][ni], false, false);
        }
    }
    // C/D layout: element r of lane -> row = 8*hi + r, col = lo
    float* P = partial + (size_t)s * B_ * NH_;
#pragma unroll
    for (int mi = 0; mi < 2; ++mi)
#pragma unroll
        for (int ni = 0; ni < 4; ++ni)
#pragma unroll
            for (int r = 0; r < 8; ++r)
                P[(size_t)(rBase + 16*mi + 8*hi + r) * NH_ + (cBase + 16*ni + lo)] =
                    acc[mi][ni][r];
}

// ===========================================================================
// h-sample: reduce split-K partials, + h_bias, sigmoid, bernoulli.
// ===========================================================================
__global__ __launch_bounds__(256) void rbm_hsample(
    const float* __restrict__ partial, const float* __restrict__ h_bias,
    float* __restrict__ p_h, float* __restrict__ h_out,
    unsigned short* __restrict__ h_bf16) {
    const int i = blockIdx.x * 256 + threadIdx.x;   // 0 .. B_*NH_-1
    float a = 0.f;
#pragma unroll
    for (int s = 0; s < SPLIT_; ++s) a += partial[(size_t)s * B_ * NH_ + i];
    a += h_bias[i & (NH_ - 1)];
    const float ph = 1.f / (1.f + __expf(-a));
    p_h[i] = ph;
    const float hv = (u01((unsigned)i, 0xA24BAED5u) < ph) ? 1.f : 0.f;
    h_out[i] = hv;
    __bf16 hb = (__bf16)hv;
    h_bf16[i] = *reinterpret_cast<unsigned short*>(&hb);
}

// ===========================================================================
// GEMM2 + fused K-softmax + bernoulli:
//   act_v[k,b,v] = sum_h h[b,h] * W[k,h,v] + v_bias[k,v]
// Reduction dim h is strided in W -> stage W tiles via LDS transposed
// ([k][v'][h'], pad 40 -> conflict-free ds_load_b128 fragment reads).
// Block: 8 waves x 16 b-rows = 128 b-rows, 32 v-cols, all 5 k in registers.
// Outputs stored with NT hint: 819 MB written once, keep L2 for W reuse.
// ===========================================================================
__global__ __launch_bounds__(256) void rbm_gemm2(
    const unsigned short* __restrict__ hmat16, const float* __restrict__ W,
    const float* __restrict__ v_bias,
    float* __restrict__ p_v, float* __restrict__ v_out) {
    __shared__ __attribute__((aligned(16))) __bf16 lds[K_][32][40];
    const __bf16* hmat = reinterpret_cast<const __bf16*>(hmat16);

    const int bx    = blockIdx.x;             // 0..7    (128 b-rows)
    const int vBase = blockIdx.y * 32;        // 0..624  (32 v-cols)
    const int lane = threadIdx.x & 31, wid = threadIdx.x >> 5;
    const int lo = lane & 15, hi = lane >> 4;
    const int rowBase = bx * 128 + wid * 16;

    v8f acc[K_][2];
#pragma unroll
    for (int k = 0; k < K_; ++k)
#pragma unroll
        for (int t = 0; t < 2; ++t)
            acc[k][t] = (v8f){0.f,0.f,0.f,0.f,0.f,0.f,0.f,0.f};

    for (int hs = 0; hs < 8; ++hs) {          // 8 x 32 = NH_ reduction
        __syncthreads();                      // protect LDS reuse
        // Cooperative coalesced stage: W[k][hs*32+r][vBase+c] -> lds[k][c][r] (bf16)
#pragma unroll
        for (int i = 0; i < 20; ++i) {
            const int idx = threadIdx.x + i * 256;     // 0..5119
            const int c = idx & 31;
            const int r = (idx >> 5) & 31;
            const int k = idx >> 10;
            lds[k][c][r] = (__bf16)W[(size_t)k * NH_ * NV_ +
                                     (size_t)(hs * 32 + r) * NV_ + (vBase + c)];
        }
        __syncthreads();

        // A fragment from bf16 h matrix (contiguous in h)
        const __bf16* hrow = hmat + (size_t)(rowBase + lo) * NH_ + hs * 32 + 16 * hi;
        v16bf a;
        {
            v8bf a0 = *(const v8bf*)(hrow);
            v8bf a1 = *(const v8bf*)(hrow + 8);
#pragma unroll
            for (int j = 0; j < 8; ++j) { a[j] = a0[j]; a[8 + j] = a1[j]; }
        }
#pragma unroll
        for (int k = 0; k < K_; ++k)
#pragma unroll
            for (int t = 0; t < 2; ++t) {
                const __bf16* bp = &lds[k][lo + 16 * t][16 * hi];
                v16bf b;
                v8bf b0 = *(const v8bf*)(bp);      // ds_load_b128
                v8bf b1 = *(const v8bf*)(bp + 8);  // ds_load_b128
#pragma unroll
                for (int j = 0; j < 8; ++j) { b[j] = b0[j]; b[8 + j] = b1[j]; }
                acc[k][t] = __builtin_amdgcn_wmma_f32_16x16x32_bf16(
                    false, a, false, b, (short)0, acc[k][t], false, false);
            }
    }

    // Fused epilogue: + v_bias, softmax over k (K_=5), bernoulli sample.
#pragma unroll
    for (int t = 0; t < 2; ++t) {
        const int col = vBase + 16 * t + lo;
        float bias[K_];
#pragma unroll
        for (int k = 0; k < K_; ++k) bias[k] = v_bias[(size_t)k * NV_ + col];
#pragma unroll
        for (int r = 0; r < 8; ++r) {
            const int row = rowBase + 8 * hi + r;
            float e[K_];
            float m = -3.0e38f;
#pragma unroll
            for (int k = 0; k < K_; ++k) {
                e[k] = acc[k][t][r] + bias[k];
                m = fmaxf(m, e[k]);
            }
            float sum = 0.f;
#pragma unroll
            for (int k = 0; k < K_; ++k) { e[k] = __expf(e[k] - m); sum += e[k]; }
            const float inv = 1.f / sum;
#pragma unroll
            for (int k = 0; k < K_; ++k) {
                const size_t off = (size_t)k * B_ * NV_ + (size_t)row * NV_ + col;
                const float p = e[k] * inv;
                __builtin_nontemporal_store(p, p_v + off);
                const float smp = (u01((unsigned)off, 0xD192ED03u) < p) ? 1.f : 0.f;
                __builtin_nontemporal_store(smp, v_out + off);
            }
        }
    }
}

// ===========================================================================
// Launch
// ===========================================================================
extern "C" void kernel_launch(void* const* d_in, const int* in_sizes, int n_in,
                              void* d_out, int out_size, void* d_ws, size_t ws_size,
                              hipStream_t stream) {
    const float* x      = (const float*)d_in[0];   // [K,B,NV]
    const float* W      = (const float*)d_in[1];   // [K,NH,NV]
    const float* v_bias = (const float*)d_in[2];   // [K,1,NV]
    const float* h_bias = (const float*)d_in[3];   // [1,NH]

    float* out  = (float*)d_out;
    float* p_h  = out;                                    // B*NH
    float* h    = out + (size_t)B_ * NH_;                 // B*NH
    float* p_v  = out + (size_t)2 * B_ * NH_;             // K*B*NV
    float* vout = p_v + (size_t)K_ * B_ * NV_;            // K*B*NV

    char* ws = (char*)d_ws;
    unsigned short* h_bf16 = (unsigned short*)ws;                 // 512 KB
    float* partial         = (float*)(ws + (1u << 20));           // 8 MB (SPLIT_*B*NH)

    dim3 g1(B_ / 128, NH_ / 128, SPLIT_);                 // 8 x 2 x 8 = 128 blocks
    rbm_gemm1<<<g1, 256, 0, stream>>>(x, W, partial);

    rbm_hsample<<<(B_ * NH_) / 256, 256, 0, stream>>>(partial, h_bias,
                                                      p_h, h, h_bf16);

    dim3 g2(B_ / 128, NV_ / 32);                          // 8 x 625 = 5000 blocks
    rbm_gemm2<<<g2, 256, 0, stream>>>(h_bf16, W, v_bias, p_v, vout);
}
